// Head_3272765079748
// MI455X (gfx1250) — matrix operand order repaired and checked by reading, not compiled, not run on hardware
//
#include <hip/hip_runtime.h>
#include <hip/hip_bf16.h>

typedef __attribute__((ext_vector_type(16))) _Float16 v16h;
typedef __attribute__((ext_vector_type(8)))  _Float16 v8h;
typedef __attribute__((ext_vector_type(8)))  float    v8f;
typedef __attribute__((ext_vector_type(4)))  float    v4f;
typedef __attribute__((ext_vector_type(4)))  int      v4i;

union AFrag {
    v16h v;
    v8h  h8[2];
    _Float16 h[16];
};

// Problem constants
#define BB 4
#define TT 4096
#define CC 1024
#define HH 64

// Async global->LDS copy (CDNA5 GLOBAL_LOAD_ASYNC_TO_LDS_B128, ASYNCcnt).
// Probe-confirmed prototype: (int4 AS(1)* gaddr, int4 AS(3)* lds, imm, cpol).
#if __has_builtin(__builtin_amdgcn_global_load_async_to_lds_b128)
#define ASYNC_COPY16(dst, src)                                              \
    __builtin_amdgcn_global_load_async_to_lds_b128(                         \
        (__attribute__((address_space(1))) v4i*)(void*)(src),               \
        (__attribute__((address_space(3))) v4i*)(void*)(dst), 0, 0)
#define ASYNC_WAIT() asm volatile("s_wait_asynccnt 0" ::: "memory")
#else
#define ASYNC_COPY16(dst, src) (*(v8h*)(dst) = *(const v8h*)(src))
#define ASYNC_WAIT() ((void)0)
#endif

// ---------------------------------------------------------------------------
// Kernel 1: convert W [C,H] f32 -> W^T [H,C] f16 (3 matrices). Fold the
// softmax scale H^-0.5 * log2(e) into Wq so scores are already in log2 units.
// ---------------------------------------------------------------------------
__global__ __launch_bounds__(256) void wconv(const float* __restrict__ Wk,
                                             const float* __restrict__ Wq,
                                             const float* __restrict__ Wv,
                                             _Float16* __restrict__ wt) {
    int idx = blockIdx.x * 256 + threadIdx.x;   // 3*1024*64 = 196608 threads
    int w = idx >> 16;                          // 65536 elems per matrix
    int r = idx & 65535;
    int c = r >> 6;
    int h = r & 63;
    const float* src = (w == 0) ? Wk : (w == 1) ? Wq : Wv;
    float scale = (w == 1) ? (0.125f * 1.44269504088896340736f) : 1.0f;
    wt[w * 65536 + h * CC + c] = (_Float16)(src[c * HH + h] * scale);
}

// ---------------------------------------------------------------------------
// Kernel 2: projections. One wave computes one 16x64 output tile of K/Q/V.
// D = X(16x1024) * W(1024x64) accumulated over 32 WMMA K-steps of 32.
// Q,K written row-major [B*T][H] f16; V written transposed [B][H][T] f16.
// ---------------------------------------------------------------------------
__global__ __launch_bounds__(128) void proj(const float* __restrict__ data,
                                            const _Float16* __restrict__ wt,
                                            _Float16* __restrict__ qh,
                                            _Float16* __restrict__ kh,
                                            _Float16* __restrict__ vt) {
    const int lane = threadIdx.x & 31;
    const int widx = threadIdx.x >> 5;
    const int gw   = blockIdx.x * 4 + widx;   // 3072 waves
    const int tile = gw / 3;                  // 0..1023 (16-row tiles of B*T)
    const int out  = gw % 3;                  // 0=K, 1=Q, 2=V
    const int ln = lane & 15;
    const int g  = (lane >> 4) & 1;

    const _Float16* w0  = wt + out * 65536;
    const float* arow   = data + (size_t)(tile * 16 + ln) * CC + g * 8;

    v8f acc[4] = {};
    for (int kt = 0; kt < 32; ++kt) {
        // A fragment: halves 0..7 at k = kt*32 + g*8, halves 8..15 at +16.
        AFrag a;
        const float* ap = arow + kt * 32;
        v4f f0 = *(const v4f*)(ap);
        v4f f1 = *(const v4f*)(ap + 4);
        v4f f2 = *(const v4f*)(ap + 16);
        v4f f3 = *(const v4f*)(ap + 20);
        for (int i = 0; i < 4; ++i) {
            a.h[i]      = (_Float16)f0[i];
            a.h[4 + i]  = (_Float16)f1[i];
            a.h[8 + i]  = (_Float16)f2[i];
            a.h[12 + i] = (_Float16)f3[i];
        }
        // B fragments from W^T: lane n = nt*16+ln, halves 0..7 at k = kt*32 + g*8, 8..15 at +16
        const _Float16* bp0 = w0 + (size_t)ln * CC + kt * 32 + g * 8;
        for (int nt = 0; nt < 4; ++nt) {
            AFrag bfr;
            const _Float16* bp = bp0 + (size_t)nt * 16 * CC;
            bfr.h8[0] = *(const v8h*)(bp);
            bfr.h8[1] = *(const v8h*)(bp + 16);
            acc[nt] = __builtin_amdgcn_wmma_f32_16x16x32_f16(
                false, a.v, false, bfr.v, (short)0, acc[nt], false, false);
        }
    }
    // D layout: lane -> n = nt*16+ln, VGPR v -> m = v + g*8
    if (out < 2) {
        _Float16* dst = (out == 1) ? qh : kh;
        for (int nt = 0; nt < 4; ++nt)
            for (int v = 0; v < 8; ++v) {
                int row = tile * 16 + v + g * 8;
                int col = nt * 16 + ln;
                dst[(size_t)row * HH + col] = (_Float16)acc[nt][v];
            }
    } else {
        int b  = tile >> 8;                       // 256 tiles per batch
        int t0 = (tile & 255) * 16 + g * 8;       // consecutive t across v
        for (int nt = 0; nt < 4; ++nt) {
            int h = nt * 16 + ln;
            AFrag pk;
            for (int v = 0; v < 8; ++v) pk.h[v] = (_Float16)acc[nt][v];
            *(v8h*)(vt + (size_t)b * (HH * TT) + (size_t)h * TT + t0) = pk.h8[0];
        }
    }
}

// ---------------------------------------------------------------------------
// Kernel 3: flash attention. Block = 4 waves = 4 consecutive q-tiles of one
// batch; K/V tiles staged cooperatively into double-buffered LDS with async
// global->LDS copies. S^T = K*Q^T so the score D-fragment IS the P A-fragment
// for the PV matmul. Online softmax in f32; per-q factors broadcast via LDS.
// ---------------------------------------------------------------------------
__global__ __launch_bounds__(128) void attn(const _Float16* __restrict__ qh,
                                            const _Float16* __restrict__ kh,
                                            const _Float16* __restrict__ vt,
                                            float* __restrict__ outp) {
    __shared__ _Float16 ldsK[2 * 32 * 64];     // 8 KB: K rows [32][64], x2
    __shared__ _Float16 ldsV[2 * 64 * 32];     // 8 KB: V^T rows [64][32], x2
    __shared__ float sbuf[64];                 // 16 floats per wave

    const int lane = threadIdx.x & 31;
    const int widx = threadIdx.x >> 5;
    const int b    = blockIdx.x >> 6;          // 64 blocks per batch
    const int i0   = (blockIdx.x & 63) * 4;    // first q-tile of this block
    const int it   = i0 + widx;                // this wave's q-tile
    const int ln = lane & 15;
    const int g  = (lane >> 4) & 1;
    float* sw = sbuf + widx * 16;

    const _Float16* kbase = kh + (size_t)(b * TT) * HH;
    const _Float16* vbase = vt + (size_t)b * (HH * TT);

    // Q as B-fragment of Q^T: lane n = q = ln; halves 0..7 at h = c*32+g*8, 8..15 at +16
    const _Float16* qbase = qh + (size_t)(b * TT + it * 16 + ln) * HH;
    AFrag qb[2];
    for (int c = 0; c < 2; ++c) {
        const _Float16* qp = qbase + c * 32 + g * 8;
        qb[c].h8[0] = *(const v8h*)(qp);
        qb[c].h8[1] = *(const v8h*)(qp + 16);
    }

    v8f o[4] = {};
    float m = -1e30f, l = 0.0f;
    const int qg = it * 16 + ln;
    const int nsteps = ((i0 + 3) >> 1) + 1;    // uniform across the block

    // Cooperative staging of one 32-key step into LDS buffer `buf`.
    auto stage = [&](int step, int buf) {
        const int sbase = step * 32;
        _Float16* lk = ldsK + buf * (32 * 64);
        _Float16* lv = ldsV + buf * (64 * 32);
        const _Float16* gk = kbase + (size_t)sbase * HH;
        const _Float16* gv = vbase + sbase;
        for (int rep = 0; rep < 2; ++rep) {
            int c = threadIdx.x + rep * 128;           // 0..255 b128 chunks
            int kr = c >> 3, ko = (c & 7) * 8;         // K: [32][64] halves
            ASYNC_COPY16(lk + kr * 64 + ko, gk + (size_t)kr * HH + ko);
            int vr = c >> 2, vo = (c & 3) * 8;         // V^T: [64][32] halves
            ASYNC_COPY16(lv + vr * 32 + vo, gv + (size_t)vr * TT + vo);
        }
    };

    stage(0, 0);
    ASYNC_WAIT();
    __syncthreads();

    for (int step = 0; step < nsteps; ++step) {
        const int buf = step & 1;
        if (step + 1 < nsteps) stage(step + 1, buf ^ 1);   // prefetch next

        const int sbase = step * 32;
        if (sbase <= it * 16 + 15) {                       // wave-uniform
            const _Float16* lk = ldsK + buf * (32 * 64);
            const _Float16* lv = ldsV + buf * (64 * 32);

            // Two 16x16 tiles of S^T (M=s, N=q), f32 accumulation over h.
            v8f st[2];
            for (int t = 0; t < 2; ++t) {
                const _Float16* kr = lk + (t * 16 + ln) * 64 + g * 8;
                v8f sa = {};
                for (int c = 0; c < 2; ++c) {
                    AFrag ka;                   // A: row s=ln, k=h
                    const _Float16* kp = kr + c * 32;
                    ka.h8[0] = *(const v8h*)(kp);
                    ka.h8[1] = *(const v8h*)(kp + 16);
                    sa = __builtin_amdgcn_wmma_f32_16x16x32_f16(
                        false, ka.v, false, qb[c].v, (short)0, sa, false, false);
                }
                st[t] = sa;
            }
            // Causal mask + online softmax partials.
            // D layout: lane -> q = ln, VGPR v -> s_local = v + g*8 per tile.
            float p0[8], p1[8];
            const bool need_mask = (sbase + 31) > (it * 16);
            float rmax = -1e30f;
            for (int v = 0; v < 8; ++v) {
                float x0 = st[0][v];
                float x1 = st[1][v];
                if (need_mask) {
                    int s0 = sbase + g * 8 + v;
                    x0 = (s0 > qg)      ? -1e30f : x0;
                    x1 = (s0 + 16 > qg) ? -1e30f : x1;
                }
                p0[v] = x0; p1[v] = x1;
                rmax = fmaxf(rmax, fmaxf(x0, x1));
            }
            rmax = fmaxf(rmax, __shfl_xor(rmax, 16, 32));
            float mnew  = fmaxf(m, rmax);
            float alpha = exp2f(m - mnew);
            float rsum  = 0.0f;
            AFrag pf;                           // P as A-fragment (16 x K=32)
            for (int v = 0; v < 8; ++v) {
                float e0 = exp2f(p0[v] - mnew);
                float e1 = exp2f(p1[v] - mnew);
                rsum += e0 + e1;
                pf.h[v]     = (_Float16)e0;     // K 0..7 / 8..15 (tile 0)
                pf.h[8 + v] = (_Float16)e1;     // K 16..23 / 24..31 (tile 1)
            }
            rsum += __shfl_xor(rsum, 16, 32);
            l = l * alpha + rsum;
            m = mnew;
            // Broadcast alpha[q] (O-accumulator rows are q = v + g*8).
            if (lane < 16) sw[ln] = alpha;
            __builtin_amdgcn_wave_barrier();
            asm volatile("s_wait_dscnt 0" ::: "memory");
            float ab[8];
            {
                const float* ap = sw + g * 8;
                for (int v = 0; v < 8; ++v) ab[v] = ap[v];
            }
            __builtin_amdgcn_wave_barrier();
            for (int ht = 0; ht < 4; ++ht)
                for (int v = 0; v < 8; ++v)
                    o[ht][v] *= ab[v];
            // O += P * V ; V^T rows in LDS give B-fragments (k=s table order).
            for (int ht = 0; ht < 4; ++ht) {
                AFrag vb;
                const _Float16* vp = lv + (ht * 16 + ln) * 32 + g * 8;
                vb.h8[0] = *(const v8h*)(vp);
                vb.h8[1] = *(const v8h*)(vp + 16);
                o[ht] = __builtin_amdgcn_wmma_f32_16x16x32_f16(
                    false, pf.v, false, vb.v, (short)0, o[ht], false, false);
            }
        }
        ASYNC_WAIT();          // next buffer fully landed in LDS
        __syncthreads();       // all waves done with current buffer
    }

    // Final 1/l normalize (broadcast like alpha) and f32 store.
    if (lane < 16) sw[ln] = 1.0f / l;
    __builtin_amdgcn_wave_barrier();
    asm volatile("s_wait_dscnt 0" ::: "memory");
    float lb[8];
    for (int v = 0; v < 8; ++v) lb[v] = sw[g * 8 + v];
    float* ob = outp + (size_t)(b * TT + it * 16) * HH;
    for (int ht = 0; ht < 4; ++ht)
        for (int v = 0; v < 8; ++v)
            ob[(size_t)(v + g * 8) * HH + ht * 16 + ln] = o[ht][v] * lb[v];
}

// ---------------------------------------------------------------------------
// Workspace layout (bytes):
//   [0)            W^T f16, 3*64*1024*2   = 393216
//   [393216)       Q  f16,  B*T*H*2       = 2097152
//   [2490368)      K  f16,  B*T*H*2       = 2097152
//   [4587520)      V^T f16, B*H*T*2       = 2097152   (total ~6.4 MB)
// ---------------------------------------------------------------------------
extern "C" void kernel_launch(void* const* d_in, const int* in_sizes, int n_in,
                              void* d_out, int out_size, void* d_ws, size_t ws_size,
                              hipStream_t stream) {
    (void)in_sizes; (void)n_in; (void)out_size; (void)ws_size;
    const float* data = (const float*)d_in[0];
    const float* Wk   = (const float*)d_in[1];
    const float* Wq   = (const float*)d_in[2];
    const float* Wv   = (const float*)d_in[3];
    float* out = (float*)d_out;

    char* ws = (char*)d_ws;
    _Float16* wt  = (_Float16*)(ws);
    _Float16* qhb = (_Float16*)(ws + 393216);
    _Float16* khb = (_Float16*)(ws + 393216 + 2097152);
    _Float16* vtb = (_Float16*)(ws + 393216 + 2 * 2097152);

    wconv<<<768, 256, 0, stream>>>(Wk, Wq, Wv, wt);          // 196608 threads
    proj <<<768, 128, 0, stream>>>(data, wt, qhb, khb, vtb); // 3072 waves
    attn <<<256, 128, 0, stream>>>(qhb, khb, vtb, out);      // 1024 waves
}
